// REINFORCE_7069516169862
// MI455X (gfx1250) — matrix operand order repaired and checked by reading, not compiled, not run on hardware
//
#include <hip/hip_runtime.h>

// ---------------------------------------------------------------------------
// Problem constants (derived from reference): H=128, D_IN=257, M=512, J=4000
// A = in_sizes[2] (400000), MAXSEL = 256.
// Strategy:
//   prep:   swizzle W0/W1 -> bf16 WMMA B-fragments, extract proc column, zero bans
//   mlp:    per-wave 16-row tile, bf16 WMMA (96 v_wmma per tile), logits -> ws
//   softmax: max-reduce, exp+sum, normalize -> probs in d_out
//   select: 256 x (masked argmax grid kernel + single-block pick/ban kernel)
// ---------------------------------------------------------------------------

typedef __attribute__((ext_vector_type(16))) __bf16 v16bf;
typedef __attribute__((ext_vector_type(8)))  float  v8f;

union Frag {
    uint4          q[2];
    unsigned short s[16];
    v16bf          v;
};

__device__ __forceinline__ unsigned short f2bf(float f) {
    unsigned int u = __float_as_uint(f);
    u += 0x7FFFu + ((u >> 16) & 1u);          // round-to-nearest-even
    return (unsigned short)(u >> 16);
}

__device__ __forceinline__ float lrelu(float v) {
    return fmaxf(v, 0.01f * v);               // leaky_relu, slope 0.01
}

// ---------------------------------------------------------------------------
// Prep: W0 [128,257] -> bf16 B-frags (8 ntiles x 8 kblocks x 32 lanes x 16),
//       W1 [128,128] -> bf16 B-frags (8 ntiles x 4 kblocks x 32 lanes x 16),
//       W0c[n] = W0[n][256], zero ban arrays.
// B-frag layout: lane l = column n%16 within ntile, element i ->
//                K = 8*(l>>4) + (i&7) + 16*(i>>3); contiguous 32B per lane.
// ---------------------------------------------------------------------------
__global__ void prep_kernel(const float* __restrict__ W0,
                            const float* __restrict__ W1,
                            unsigned short* __restrict__ w0b,
                            unsigned short* __restrict__ w1b,
                            float* __restrict__ w0c,
                            int* __restrict__ banm,
                            int* __restrict__ banj) {
    int t = blockIdx.x * 256 + threadIdx.x;
    if (t < 32768) {                                   // W0B
        int i = t & 15, lane = (t >> 4) & 31, r = t >> 9;
        int kb = r & 7, nt = r >> 3;
        int n = nt * 16 + (lane & 15);
        int k = kb * 32 + 8 * (lane >> 4) + (i & 7) + 16 * (i >> 3);
        w0b[t] = f2bf(W0[n * 257 + k]);
    } else if (t < 49152) {                            // W1B
        int u = t - 32768;
        int i = u & 15, lane = (u >> 4) & 31, r = u >> 9;
        int kb = r & 3, nt = r >> 2;
        int n = nt * 16 + (lane & 15);
        int k = kb * 32 + 8 * (lane >> 4) + (i & 7) + 16 * (i >> 3);
        w1b[u] = f2bf(W1[n * 128 + k]);
    } else if (t < 49280) {                            // proc-time column of W0
        int n = t - 49152;
        w0c[n] = W0[n * 257 + 256];
    } else if (t < 49792) {                            // ban_machine[512]
        banm[t - 49280] = 0;
    } else if (t < 53792) {                            // ban_job[4000]
        banj[t - 49792] = 0;
    }
}

// ---------------------------------------------------------------------------
// MLP: 256 threads = 8 waves/block, each wave owns a 16-candidate tile.
// Layer0: K=256 (8 kblocks), +rank-1 proc term in acc init. Layer1: K=128.
// ---------------------------------------------------------------------------
__global__ __launch_bounds__(256) void mlp_kernel(
    const float* __restrict__ x_m, const float* __restrict__ x_op,
    const float* __restrict__ proc,
    const float* __restrict__ b0v, const float* __restrict__ b1v,
    const float* __restrict__ W2, const float* __restrict__ b2v,
    const int* __restrict__ m_ids, const int* __restrict__ op_idxs,
    const float* __restrict__ w0c,
    const unsigned short* __restrict__ w0b,
    const unsigned short* __restrict__ w1b,
    float* __restrict__ logits, int A) {

    __shared__ __align__(16) unsigned short h0s[8][16][136];  // padded stride

    const int wave = threadIdx.x >> 5;
    const int lane = threadIdx.x & 31;
    const int ln   = lane & 15;         // row (A-frag) / column (B,C frags)
    const int hl   = lane >> 4;         // half selector
    const int tileBase = (blockIdx.x * 8 + wave) * 16;

    int aRow = tileBase + ln; if (aRow >= A) aRow = A - 1;
    const int mid = m_ids[aRow];
    const int oid = op_idxs[aRow];
    const float* __restrict__ mrow = x_m  + (size_t)mid * 128;
    const float* __restrict__ orow = x_op + (size_t)oid * 128;

    // ---- layer 0 accumulators: C[M=i+8*hl][N=nt*16+ln], init = b0 + pt*W0c
    v8f acc[8];
    {
        float ptv[8];
#pragma unroll
        for (int i = 0; i < 8; ++i) {
            int r = tileBase + 8 * hl + i; if (r >= A) r = A - 1;
            ptv[i] = proc[r];
        }
#pragma unroll
        for (int nt = 0; nt < 8; ++nt) {
            const int n = nt * 16 + ln;
            const float bn = b0v[n];
            const float wc = w0c[n];
#pragma unroll
            for (int i = 0; i < 8; ++i) acc[nt][i] = fmaf(ptv[i], wc, bn);
        }
    }

    // ---- layer 0: 8 kblocks x 8 ntiles = 64 WMMAs
#pragma unroll
    for (int kb = 0; kb < 8; ++kb) {
        const float* rowp = (kb < 4) ? (mrow + kb * 32) : (orow + (kb - 4) * 32);
        const float4* p0 = reinterpret_cast<const float4*>(rowp + 8 * hl);
        const float4* p1 = reinterpret_cast<const float4*>(rowp + 16 + 8 * hl);
        float4 q0 = p0[0], q1 = p0[1], q2 = p1[0], q3 = p1[1];
        const float fv[16] = {q0.x,q0.y,q0.z,q0.w, q1.x,q1.y,q1.z,q1.w,
                              q2.x,q2.y,q2.z,q2.w, q3.x,q3.y,q3.z,q3.w};
        Frag af;
#pragma unroll
        for (int j = 0; j < 16; ++j) af.s[j] = f2bf(fv[j]);
#pragma unroll
        for (int nt = 0; nt < 8; ++nt) {
            Frag bf;
            const uint4* bp = reinterpret_cast<const uint4*>(
                w0b + ((nt * 8 + kb) * 32 + lane) * 16);
            bf.q[0] = bp[0]; bf.q[1] = bp[1];
            acc[nt] = __builtin_amdgcn_wmma_f32_16x16x32_bf16(
                false, af.v, false, bf.v, (short)0, acc[nt], false, false);
        }
    }

    // ---- leaky + C-layout -> LDS (padded rows kill bank conflicts)
#pragma unroll
    for (int nt = 0; nt < 8; ++nt) {
        const int col = nt * 16 + ln;
#pragma unroll
        for (int i = 0; i < 8; ++i)
            h0s[wave][8 * hl + i][col] = f2bf(lrelu(acc[nt][i]));
    }
    __syncthreads();

    // ---- layer 1: 4 kblocks x 8 ntiles = 32 WMMAs
    v8f acc2[8];
#pragma unroll
    for (int nt = 0; nt < 8; ++nt) {
        const float bn = b1v[nt * 16 + ln];
#pragma unroll
        for (int i = 0; i < 8; ++i) acc2[nt][i] = bn;
    }
    const unsigned short* hrow = &h0s[wave][ln][0];
#pragma unroll
    for (int kb = 0; kb < 4; ++kb) {
        Frag af;
        af.q[0] = *reinterpret_cast<const uint4*>(hrow + kb * 32 + 8 * hl);
        af.q[1] = *reinterpret_cast<const uint4*>(hrow + kb * 32 + 16 + 8 * hl);
#pragma unroll
        for (int nt = 0; nt < 8; ++nt) {
            Frag bf;
            const uint4* bp = reinterpret_cast<const uint4*>(
                w1b + ((nt * 4 + kb) * 32 + lane) * 16);
            bf.q[0] = bp[0]; bf.q[1] = bp[1];
            acc2[nt] = __builtin_amdgcn_wmma_f32_16x16x32_bf16(
                false, af.v, false, bf.v, (short)0, acc2[nt], false, false);
        }
    }

    // ---- output layer: logits[m] = sum_n lrelu(h1[m,n]) * W2[n] + b2
    float t[8];
#pragma unroll
    for (int i = 0; i < 8; ++i) t[i] = 0.f;
#pragma unroll
    for (int nt = 0; nt < 8; ++nt) {
        const float w = W2[nt * 16 + ln];
#pragma unroll
        for (int i = 0; i < 8; ++i) t[i] = fmaf(lrelu(acc2[nt][i]), w, t[i]);
    }
#pragma unroll
    for (int m = 1; m <= 8; m <<= 1) {
#pragma unroll
        for (int i = 0; i < 8; ++i) t[i] += __shfl_xor(t[i], m, 32);
    }
    if (ln < 8) {
        const int row = tileBase + 8 * hl + ln;
        if (row < A) logits[row] = t[ln] + b2v[0];
    }
}

// ---------------------------------------------------------------------------
// Softmax helpers
// ---------------------------------------------------------------------------
__global__ __launch_bounds__(256) void red_max(const float* __restrict__ x,
                                               float* __restrict__ out, int A) {
    __shared__ float sv[256];
    int t = threadIdx.x, base = blockIdx.x * 512 + t;
    float v = -INFINITY;
    if (base < A)       v = x[base];
    if (base + 256 < A) v = fmaxf(v, x[base + 256]);
    sv[t] = v; __syncthreads();
    for (int o = 128; o > 0; o >>= 1) {
        if (t < o) sv[t] = fmaxf(sv[t], sv[t + o]);
        __syncthreads();
    }
    if (t == 0) out[blockIdx.x] = sv[0];
}

__global__ __launch_bounds__(256) void red_max_fin(const float* __restrict__ in,
                                                   int n, float* __restrict__ g) {
    __shared__ float sv[256];
    int t = threadIdx.x;
    float v = -INFINITY;
    for (int i = t; i < n; i += 256) v = fmaxf(v, in[i]);
    sv[t] = v; __syncthreads();
    for (int o = 128; o > 0; o >>= 1) {
        if (t < o) sv[t] = fmaxf(sv[t], sv[t + o]);
        __syncthreads();
    }
    if (t == 0) *g = sv[0];
}

__global__ __launch_bounds__(256) void exp_sum(const float* __restrict__ logits,
                                               const float* __restrict__ gmax,
                                               float* __restrict__ probs,
                                               float* __restrict__ out, int A) {
    __shared__ float sv[256];
    int t = threadIdx.x, base = blockIdx.x * 512 + t;
    float M = *gmax, s = 0.f;
    if (base < A)       { float e = expf(logits[base] - M);       probs[base] = e;       s += e; }
    if (base + 256 < A) { float e = expf(logits[base + 256] - M); probs[base + 256] = e; s += e; }
    sv[t] = s; __syncthreads();
    for (int o = 128; o > 0; o >>= 1) {
        if (t < o) sv[t] += sv[t + o];
        __syncthreads();
    }
    if (t == 0) out[blockIdx.x] = sv[0];
}

__global__ __launch_bounds__(256) void red_sum_fin(const float* __restrict__ in,
                                                   int n, float* __restrict__ g) {
    __shared__ float sv[256];
    int t = threadIdx.x;
    float v = 0.f;
    for (int i = t; i < n; i += 256) v += in[i];
    sv[t] = v; __syncthreads();
    for (int o = 128; o > 0; o >>= 1) {
        if (t < o) sv[t] += sv[t + o];
        __syncthreads();
    }
    if (t == 0) *g = sv[0];
}

__global__ __launch_bounds__(256) void normalize_k(float* __restrict__ probs,
                                                   const float* __restrict__ gsum,
                                                   int A) {
    int i = blockIdx.x * 256 + threadIdx.x;
    if (i < A) probs[i] = probs[i] / (*gsum);
}

// ---------------------------------------------------------------------------
// Greedy selection: masked argmax (ties -> lowest index, like jnp.argmax)
// ---------------------------------------------------------------------------
__global__ __launch_bounds__(256) void sel_scan(
    const float* __restrict__ probs, const int* __restrict__ jobs,
    const int* __restrict__ ms, const int* __restrict__ banj,
    const int* __restrict__ banm, float* __restrict__ selv,
    int* __restrict__ seli, int A) {
    __shared__ float sv[256];
    __shared__ int   si[256];
    int t = threadIdx.x, base = blockIdx.x * 1024 + t;
    float bv = -INFINITY; int bi = 0x7fffffff;
#pragma unroll
    for (int k = 0; k < 4; ++k) {
        int i = base + k * 256;
        if (i < A) {
            float p = probs[i];
            bool ok = (p > 0.f) && (banj[jobs[i]] == 0) && (banm[ms[i]] == 0);
            if (ok && (p > bv || (p == bv && i < bi))) { bv = p; bi = i; }
        }
    }
    sv[t] = bv; si[t] = bi; __syncthreads();
    for (int o = 128; o > 0; o >>= 1) {
        if (t < o) {
            float ov = sv[t + o]; int oi = si[t + o];
            if (ov > sv[t] || (ov == sv[t] && oi < si[t])) { sv[t] = ov; si[t] = oi; }
        }
        __syncthreads();
    }
    if (t == 0) { selv[blockIdx.x] = sv[0]; seli[blockIdx.x] = si[0]; }
}

__global__ __launch_bounds__(256) void sel_pick(
    const float* __restrict__ selv, const int* __restrict__ seli, int nblk,
    const int* __restrict__ jobs, const int* __restrict__ ms,
    int* __restrict__ banj, int* __restrict__ banm,
    int* __restrict__ selected, int s) {
    __shared__ float sv[256];
    __shared__ int   si[256];
    int t = threadIdx.x;
    float bv = -INFINITY; int bi = 0x7fffffff;
    for (int i = t; i < nblk; i += 256) {
        float v = selv[i]; int idx = seli[i];
        if (v > bv || (v == bv && idx < bi)) { bv = v; bi = idx; }
    }
    sv[t] = bv; si[t] = bi; __syncthreads();
    for (int o = 128; o > 0; o >>= 1) {
        if (t < o) {
            float ov = sv[t + o]; int oi = si[t + o];
            if (ov > sv[t] || (ov == sv[t] && oi < si[t])) { sv[t] = ov; si[t] = oi; }
        }
        __syncthreads();
    }
    if (t == 0) {
        float v = sv[0]; int idx = si[0];
        if (v > 0.f) {                    // something valid existed
            selected[s] = idx;
            banj[jobs[idx]] = 1;
            banm[ms[idx]]   = 1;
        } else {
            selected[s] = -1;
        }
    }
}

// ---------------------------------------------------------------------------
extern "C" void kernel_launch(void* const* d_in, const int* in_sizes, int n_in,
                              void* d_out, int out_size, void* d_ws, size_t ws_size,
                              hipStream_t stream) {
    (void)n_in; (void)out_size; (void)ws_size;
    const float* x_m    = (const float*)d_in[0];
    const float* x_op   = (const float*)d_in[1];
    const float* proc   = (const float*)d_in[2];
    const float* W0     = (const float*)d_in[3];
    const float* b0v    = (const float*)d_in[4];
    const float* W1     = (const float*)d_in[5];
    const float* b1v    = (const float*)d_in[6];
    const float* W2     = (const float*)d_in[7];
    const float* b2v    = (const float*)d_in[8];
    const int*   m_ids  = (const int*)d_in[9];
    const int*   op_idx = (const int*)d_in[10];
    const int*   job_id = (const int*)d_in[11];
    // d_in[12] = max_select (device scalar, == 256); loop count must be static
    // for graph capture, so use the reference constant.
    const int A = in_sizes[2];
    const int MAXSEL = 256;

    float* probs    = (float*)d_out;
    int*   selected = ((int*)d_out) + A;

    // workspace layout (floats). w0b offset = (A + 8740)*4 bytes, 16B aligned.
    float* ws     = (float*)d_ws;
    float* logits = ws;                       // [A]
    float* redA   = ws + A;                   // [1024]
    float* redB   = redA + 1024;              // [1024]
    float* gmax   = redB + 1024;              // [1] (+3 pad)
    float* gsum   = gmax + 1;
    float* selv   = redB + 1024 + 4;          // [1024]
    int*   seli   = (int*)(selv + 1024);      // [1024]
    int*   banm   = seli + 1024;              // [512]
    int*   banj   = banm + 512;               // [4000]
    float* w0c    = (float*)(banj + 4000);    // [128]
    unsigned short* w0b = (unsigned short*)(w0c + 128);  // [32768] bf16
    unsigned short* w1b = w0b + 32768;                   // [16384] bf16

    prep_kernel<<<211, 256, 0, stream>>>(W0, W1, w0b, w1b, w0c, banm, banj);

    int mlpBlocks = (A + 127) / 128;
    mlp_kernel<<<mlpBlocks, 256, 0, stream>>>(x_m, x_op, proc, b0v, b1v, W2, b2v,
                                              m_ids, op_idx, w0c, w0b, w1b,
                                              logits, A);

    int nred = (A + 511) / 512;   // 782
    red_max<<<nred, 256, 0, stream>>>(logits, redA, A);
    red_max_fin<<<1, 256, 0, stream>>>(redA, nred, gmax);
    exp_sum<<<nred, 256, 0, stream>>>(logits, gmax, probs, redB, A);
    red_sum_fin<<<1, 256, 0, stream>>>(redB, nred, gsum);
    normalize_k<<<(A + 255) / 256, 256, 0, stream>>>(probs, gsum, A);

    int selB = (A + 1023) / 1024; // 391
    for (int s = 0; s < MAXSEL; ++s) {
        sel_scan<<<selB, 256, 0, stream>>>(probs, job_id, m_ids, banj, banm,
                                           selv, seli, A);
        sel_pick<<<1, 256, 0, stream>>>(selv, seli, selB, job_id, m_ids,
                                        banj, banm, selected, s);
    }
}
